// FractalAttention_71004399337603
// MI455X (gfx1250) — matrix operand order repaired and checked by reading, not compiled, not run on hardware
//
#include <hip/hip_runtime.h>
#include <hip/hip_bf16.h>

// ---------------------------------------------------------------------------
// FractalAttention for MI455X (gfx1250, wave32, WMMA + TDM)
//   B=2, S=2048, D=1024, H=16, HD=64, K(clusters)=4
//   ~70 GFLOP dense -> bf16 WMMA path; ~1GB mean_aw RMW traffic -> keep
//   16x2048 score tile in 320KB WGP LDS, single RMW stream per head.
//   A-operand of the projection GEMMs staged via Tensor Data Mover (TENSORcnt);
//   V-operand of P@V gathered with per-element strided global loads (stride HDn).
// ---------------------------------------------------------------------------

#define Bn 2
#define Sn 2048
#define Dn 1024
#define Hn 16
#define HDn 64
#define NC 4
#define TOKENS (Bn * Sn)          // 4096

typedef __attribute__((ext_vector_type(16))) __bf16 v16bf;
typedef __attribute__((ext_vector_type(8)))  float  v8f;
typedef __attribute__((ext_vector_type(4)))  unsigned int v4u;
typedef __attribute__((ext_vector_type(8)))  int v8i;
typedef __attribute__((ext_vector_type(4)))  int v4i;

__device__ __forceinline__ float waveMax(float v) {
  #pragma unroll
  for (int o = 16; o > 0; o >>= 1) v = fmaxf(v, __shfl_xor(v, o, 32));
  return v;
}
__device__ __forceinline__ float waveSum(float v) {
  #pragma unroll
  for (int o = 16; o > 0; o >>= 1) v += __shfl_xor(v, o, 32);
  return v;
}

// ---------------------------------------------------------------------------
// 0) zero-init (mean_aw region of d_out is accumulated into)
// ---------------------------------------------------------------------------
__global__ void zero_f32(float* p, int n) {
  int i = blockIdx.x * blockDim.x + threadIdx.x;
  if (i < n) p[i] = 0.0f;
}

// ---------------------------------------------------------------------------
// 1) fp32 -> bf16 conversion (weights)
// ---------------------------------------------------------------------------
__global__ void cvt_bf16(const float* __restrict__ src, __bf16* __restrict__ dst, int n) {
  int i = blockIdx.x * blockDim.x + threadIdx.x;
  if (i < n) dst[i] = (__bf16)src[i];
}

// ---------------------------------------------------------------------------
// 2) embedding gather + fractal-cluster perturbation, one block per token.
// ---------------------------------------------------------------------------
__global__ void embed_cluster(const int* __restrict__ X,
                              const float* __restrict__ emb,
                              const float* __restrict__ centers,
                              float* __restrict__ e_f32,
                              __bf16* __restrict__ e_bf16) {
  const int tok = blockIdx.x;               // 0..4095
  const int tid = threadIdx.x;              // 256 threads, 8 waves
  const int lane = tid & 31, wave = tid >> 5;

  const int id = X[tok];
  const float* er = emb + (size_t)id * Dn;

  float ex[4];
  float ee = 0.0f, dot[NC] = {0, 0, 0, 0}, cc[NC] = {0, 0, 0, 0};
  #pragma unroll
  for (int i = 0; i < 4; ++i) {
    const int d = tid + i * 256;
    const float v = er[d];
    ex[i] = v;
    ee += v * v;
    #pragma unroll
    for (int c = 0; c < NC; ++c) {
      const float cv = centers[c * Dn + d];
      dot[c] += v * cv;
      cc[c] += cv * cv;
    }
  }
  __shared__ float s_red[8][1 + 2 * NC];
  ee = waveSum(ee);
  #pragma unroll
  for (int c = 0; c < NC; ++c) { dot[c] = waveSum(dot[c]); cc[c] = waveSum(cc[c]); }
  if (lane == 0) {
    s_red[wave][0] = ee;
    #pragma unroll
    for (int c = 0; c < NC; ++c) { s_red[wave][1 + c] = dot[c]; s_red[wave][1 + NC + c] = cc[c]; }
  }
  __syncthreads();

  __shared__ float s_cw[NC];
  if (tid == 0) {
    float E = 0.0f, Dt[NC] = {0, 0, 0, 0}, Cc[NC] = {0, 0, 0, 0};
    for (int w = 0; w < 8; ++w) {
      E += s_red[w][0];
      #pragma unroll
      for (int c = 0; c < NC; ++c) { Dt[c] += s_red[w][1 + c]; Cc[c] += s_red[w][1 + NC + c]; }
    }
    float logit[NC], mx = -3.0e38f;
    #pragma unroll
    for (int c = 0; c < NC; ++c) {
      const float d2 = fmaxf(E + Cc[c] - 2.0f * Dt[c], 0.0f);
      logit[c] = -sqrtf(d2);          // FSF = 1.0
      mx = fmaxf(mx, logit[c]);
    }
    float sum = 0.0f;
    #pragma unroll
    for (int c = 0; c < NC; ++c) { logit[c] = __expf(logit[c] - mx); sum += logit[c]; }
    const float inv = 1.0f / sum;
    #pragma unroll
    for (int c = 0; c < NC; ++c) s_cw[c] = logit[c] * inv;
  }
  __syncthreads();

  #pragma unroll
  for (int i = 0; i < 4; ++i) {
    const int d = tid + i * 256;
    float add = 0.0f;
    #pragma unroll
    for (int c = 0; c < NC; ++c) add += s_cw[c] * centers[c * Dn + d];
    const float v = ex[i] + 0.1f * add;
    e_f32[(size_t)tok * Dn + d] = v;
    e_bf16[(size_t)tok * Dn + d] = (__bf16)v;
  }
}

// ---------------------------------------------------------------------------
// 3) WMMA bf16 GEMM:  C[m,n] = sum_k A[m,k] * W[n,k]   (y = x @ W^T)
//    A tile (32 rows x 1024 K = 64KB) staged into LDS by the Tensor Data
//    Mover (one tensor_load_to_lds per block, TENSORcnt-tracked); W streamed
//    from global with b128 loads + prefetch. One 16x16 C tile per wave.
// ---------------------------------------------------------------------------
__global__ void __launch_bounds__(256)
gemm_wmma_bf16(const __bf16* __restrict__ A, const __bf16* __restrict__ W,
               float* __restrict__ Cf, __bf16* __restrict__ Cb, int headed) {
  const int Kd = Dn, N = Dn;
  const int lane = threadIdx.x & 31, wave = threadIdx.x >> 5;
  const int half = lane >> 4, lrow = lane & 15;
  const int mtile = blockIdx.y * 32 + (wave >> 2) * 16;
  const int ntile = blockIdx.x * 64 + (wave & 3) * 16;

  __shared__ __bf16 Asmem[32 * Dn];           // 64 KB of the 320 KB WGP pool
  const __bf16* Ablk = A + (size_t)(blockIdx.y * 32) * Kd;

#if __has_builtin(__builtin_amdgcn_tensor_load_to_lds) && __has_builtin(__builtin_amdgcn_s_wait_tensorcnt)
  if (threadIdx.x < 32) {
    // D# per CDNA5 ISA: 2D tensor (1024 x 4096 bf16),
    // tile 1024 x 32, row stride 1024 elements, dest = Asmem.
    const unsigned long long ga = (unsigned long long)(uintptr_t)Ablk;
    const unsigned ldsOff = (unsigned)(uintptr_t)Asmem;     // addr[31:0] = LDS byte offset
    v4u g0 = { 1u,                                          // count=1, user mode
               ldsOff,                                      // lds_addr
               (unsigned)ga,                                // global_addr[31:0]
               ((unsigned)(ga >> 32) & 0x01ffffffu) | (2u << 30) };  // addr[56:32], type=2
    v8i g1 = { (int)(1u << 16),          // data_size=1 (2 bytes); no mask/flags
               (int)(1024u << 16),       // tensor_dim0[15:0] = 1024 (K elements)
               (int)(4096u << 16),       // tensor_dim0 hi = 0, tensor_dim1[15:0] = 4096 rows
               (int)(1024u << 16),       // tensor_dim1 hi = 0, tile_dim0 = 1024
               32,                       // tile_dim1 = 32 rows, tile_dim2 = 0
               1024,                     // tensor_dim0_stride[31:0] = 1024 elements
               0, 0 };
    v4i gz = { 0, 0, 0, 0 };
    v8i gz8 = { 0, 0, 0, 0, 0, 0, 0, 0 };
    // 6-arg form on this toolchain: (g0, g1, g2, g3, g4, cpol)
    __builtin_amdgcn_tensor_load_to_lds(g0, g1, gz, gz, gz8, 0);
    __builtin_amdgcn_s_wait_tensorcnt(0);
  }
#else
  // Fallback: cooperative copy (no TDM builtin on this toolchain)
  for (int i = threadIdx.x; i < 32 * Dn / 8; i += 256)
    ((uint4*)Asmem)[i] = ((const uint4*)Ablk)[i];
#endif
  __syncthreads();

  const __bf16* aPtr = Asmem + (size_t)((wave >> 2) * 16 + lrow) * Kd + half * 16;
  const __bf16* wPtr = W + (size_t)(ntile + lrow) * Kd + half * 16;

  v8f acc = {};
  for (int k0 = 0; k0 < Kd; k0 += 32) {
    const v16bf af = *(const v16bf*)(aPtr + k0);           // ds_load path
    const v16bf wf = *(const v16bf*)(wPtr + k0);           // global b128 path
    __builtin_prefetch(wPtr + k0 + 128, 0, 1);             // prefetch W stream
    acc = __builtin_amdgcn_wmma_f32_16x16x32_bf16(
        false, af, false, wf, (short)0, acc, false, false);
  }

  #pragma unroll
  for (int v = 0; v < 8; ++v) {
    const int m = mtile + half * 8 + v;       // C layout: lane n, VGPR v -> M
    const int n = ntile + lrow;
    if (!headed) {
      Cf[(size_t)m * N + n] = acc[v];
    } else {
      const int b = m >> 11, s = m & (Sn - 1);
      const int h = n >> 6, hd = n & (HDn - 1);
      Cb[(((size_t)b * Hn + h) * Sn + s) * HDn + hd] = (__bf16)acc[v];
    }
  }
}

// ---------------------------------------------------------------------------
// 4) Attention: one block per (b, 16-query tile); loop over 16 heads.
//    Phase 1: scores = Q K^T / 8 via WMMA -> 16x2048 LDS tile (128 KB).
//    Phase 2: per-row softmax (wave reductions), fold prob/16 into global
//             head-mean (single-owner RMW, no atomics needed).
//    Phase 3: P @ V via WMMA; V's B-operand is column-major in memory, so
//             gather it element by element (stride HDn) with plain global loads.
//    160 KB dynamic LDS -- requires CDNA5's 320 KB WGP LDS pool.
// ---------------------------------------------------------------------------
__global__ void __launch_bounds__(256, 1)
attention_wmma(const __bf16* __restrict__ Q, const __bf16* __restrict__ Kb,
               const __bf16* __restrict__ V, __bf16* __restrict__ attn_out,
               float* __restrict__ mean_out) {
  extern __shared__ float smem[];
  float* sc = smem;                   // [16][2048] scores/probs
  float* pv = smem + 16 * Sn;         // [8][16][64] per-wave PV partials

  const int b = blockIdx.x >> 7;
  const int qt = blockIdx.x & 127;
  const int tid = threadIdx.x, lane = tid & 31, wave = tid >> 5;
  const int half = lane >> 4, lrow = lane & 15;

  for (int h = 0; h < Hn; ++h) {
    const __bf16* q = Q + (((size_t)b * Hn + h) * Sn + qt * 16) * HDn;
    const __bf16* k = Kb + (((size_t)b * Hn + h) * Sn) * HDn;
    const __bf16* v = V + (((size_t)b * Hn + h) * Sn) * HDn;

    // ---- Phase 1: scores -> LDS -----------------------------------------
    for (int kt = wave; kt < Sn / 16; kt += 8) {
      v8f acc = {};
      #pragma unroll
      for (int d0 = 0; d0 < HDn; d0 += 32) {
        const v16bf aq = *(const v16bf*)(q + (size_t)lrow * HDn + d0 + half * 16);
        const v16bf bk = *(const v16bf*)(k + (size_t)(kt * 16 + lrow) * HDn + d0 + half * 16);
        acc = __builtin_amdgcn_wmma_f32_16x16x32_bf16(
            false, aq, false, bk, (short)0, acc, false, false);
      }
      #pragma unroll
      for (int vv = 0; vv < 8; ++vv)
        sc[(half * 8 + vv) * Sn + kt * 16 + lrow] = acc[vv] * 0.125f;  // 1/sqrt(64), FSF=1
    }
    __syncthreads();

    // ---- Phase 2: softmax + head-mean accumulation ----------------------
    #pragma unroll
    for (int rr = 0; rr < 2; ++rr) {
      const int r = wave * 2 + rr;
      float mx = -3.0e38f;
      for (int i = lane; i < Sn; i += 32) mx = fmaxf(mx, sc[r * Sn + i]);
      mx = waveMax(mx);
      float ssum = 0.0f;
      for (int i = lane; i < Sn; i += 32) {
        const float p = __expf(sc[r * Sn + i] - mx);
        sc[r * Sn + i] = p;
        ssum += p;
      }
      ssum = waveSum(ssum);
      const float inv = 1.0f / ssum;
      float* mrow = mean_out + ((size_t)b * Sn + qt * 16 + r) * Sn;
      for (int i = lane; i < Sn; i += 32) {
        const float p = sc[r * Sn + i] * inv;
        sc[r * Sn + i] = p;
        mrow[i] += p * (1.0f / Hn);   // block owns these rows: plain RMW
      }
    }
    __syncthreads();

    // ---- Phase 3: P @ V --------------------------------------------------
    v8f acc[4] = {{}, {}, {}, {}};
    for (int kc = wave; kc < Sn / 32; kc += 8) {
      v16bf ap;
      #pragma unroll
      for (int j = 0; j < 16; ++j)
        ap[j] = (__bf16)sc[lrow * Sn + kc * 32 + half * 16 + j];
      #pragma unroll
      for (int nt = 0; nt < 4; ++nt) {
        // V tile rows kk = kc*32 .. kc*32+31, cols nt*16 .. nt*16+15.
        // B operand: lane column n = lrow, element j -> k = half*16 + j.
        union { v4u r[2]; v16bf f; } bu;
        #pragma unroll
        for (int j = 0; j < 16; ++j)
          bu.f[j] = v[((size_t)kc * 32 + half * 16 + j) * HDn + nt * 16 + lrow];
        acc[nt] = __builtin_amdgcn_wmma_f32_16x16x32_bf16(
            false, ap, false, bu.f, (short)0, acc[nt], false, false);
      }
    }
    #pragma unroll
    for (int nt = 0; nt < 4; ++nt)
      #pragma unroll
      for (int vv = 0; vv < 8; ++vv)
        pv[wave * 1024 + (half * 8 + vv) * HDn + nt * 16 + lrow] = acc[nt][vv];
    __syncthreads();

    for (int idx = tid; idx < 16 * HDn; idx += 256) {
      float s = 0.0f;
      #pragma unroll
      for (int w = 0; w < 8; ++w) s += pv[w * 1024 + idx];
      const int qr = idx >> 6, dd = idx & (HDn - 1);
      attn_out[((size_t)b * Sn + qt * 16 + qr) * Dn + h * HDn + dd] = (__bf16)s;
    }
    __syncthreads();   // protect sc/pv before next head
  }
}

// ---------------------------------------------------------------------------
// 5) residual + bias + LayerNorm, one block per token
// ---------------------------------------------------------------------------
__global__ void resid_layernorm(const float* __restrict__ e,
                                const float* __restrict__ proj,
                                const float* __restrict__ bo,
                                const float* __restrict__ gamma,
                                const float* __restrict__ beta,
                                float* __restrict__ out) {
  const int tok = blockIdx.x;
  const int tid = threadIdx.x, lane = tid & 31, wave = tid >> 5;
  __shared__ float sr[8], sr2[8];

  float x[4];
  float s = 0.0f;
  #pragma unroll
  for (int i = 0; i < 4; ++i) {
    const int d = tid + i * 256;
    x[i] = e[(size_t)tok * Dn + d] + proj[(size_t)tok * Dn + d] + bo[d];
    s += x[i];
  }
  s = waveSum(s);
  if (lane == 0) sr[wave] = s;
  __syncthreads();
  float mu = 0.0f;
  #pragma unroll
  for (int w = 0; w < 8; ++w) mu += sr[w];
  mu *= (1.0f / Dn);

  float vs = 0.0f;
  #pragma unroll
  for (int i = 0; i < 4; ++i) { const float t = x[i] - mu; vs += t * t; }
  vs = waveSum(vs);
  if (lane == 0) sr2[wave] = vs;
  __syncthreads();
  float var = 0.0f;
  #pragma unroll
  for (int w = 0; w < 8; ++w) var += sr2[w];
  var *= (1.0f / Dn);
  const float rstd = rsqrtf(var + 1e-5f);

  #pragma unroll
  for (int i = 0; i < 4; ++i) {
    const int d = tid + i * 256;
    out[(size_t)tok * Dn + d] = (x[i] - mu) * rstd * gamma[d] + beta[d];
  }
}

// ---------------------------------------------------------------------------
// launch
// ---------------------------------------------------------------------------
extern "C" void kernel_launch(void* const* d_in, const int* in_sizes, int n_in,
                              void* d_out, int out_size, void* d_ws, size_t ws_size,
                              hipStream_t stream) {
  const int*   X       = (const int*)d_in[0];
  const float* emb     = (const float*)d_in[1];
  const float* centers = (const float*)d_in[2];
  const float* wq      = (const float*)d_in[3];
  const float* wk      = (const float*)d_in[4];
  const float* wv      = (const float*)d_in[5];
  const float* wo      = (const float*)d_in[6];
  const float* bo      = (const float*)d_in[7];
  const float* gamma   = (const float*)d_in[8];
  const float* beta    = (const float*)d_in[9];

  float* out      = (float*)d_out;                       // (B,S,D) f32
  float* mean_out = out + (size_t)Bn * Sn * Dn;          // (B,S,S) f32

  // workspace layout (80 MB total)
  char* ws = (char*)d_ws;
  float*  e_f32  = (float*)(ws);                         // 16 MB
  __bf16* e_b    = (__bf16*)(ws + (16u << 20));          //  8 MB
  __bf16* wq_b   = (__bf16*)(ws + (24u << 20));          //  2 MB
  __bf16* wk_b   = wq_b + (1u << 20);
  __bf16* wv_b   = wk_b + (1u << 20);
  __bf16* wo_b   = wv_b + (1u << 20);
  __bf16* q_b    = (__bf16*)(ws + (32u << 20));          //  8 MB (B,H,S,HD)
  __bf16* k_b    = (__bf16*)(ws + (40u << 20));          //  8 MB
  __bf16* v_b    = (__bf16*)(ws + (48u << 20));          //  8 MB
  __bf16* attn_b = (__bf16*)(ws + (56u << 20));          //  8 MB (B,S,D)
  float*  proj   = (float*)(ws + (64u << 20));           // 16 MB

  // 0) zero the accumulated mean_aw region of d_out
  const int nMean = Bn * Sn * Sn;
  zero_f32<<<(nMean + 255) / 256, 256, 0, stream>>>(mean_out, nMean);

  // 1) weights -> bf16
  const int nW = Dn * Dn;
  cvt_bf16<<<(nW + 255) / 256, 256, 0, stream>>>(wq, wq_b, nW);
  cvt_bf16<<<(nW + 255) / 256, 256, 0, stream>>>(wk, wk_b, nW);
  cvt_bf16<<<(nW + 255) / 256, 256, 0, stream>>>(wv, wv_b, nW);
  cvt_bf16<<<(nW + 255) / 256, 256, 0, stream>>>(wo, wo_b, nW);

  // 2) embedding + cluster perturbation
  embed_cluster<<<TOKENS, 256, 0, stream>>>(X, emb, centers, e_f32, e_b);

  // 3) Q/K/V projections (WMMA + TDM-staged A), headed bf16 output
  dim3 gGemm(Dn / 64, TOKENS / 32);   // (16, 128)
  gemm_wmma_bf16<<<gGemm, 256, 0, stream>>>(e_b, wq_b, nullptr, q_b, 1);
  gemm_wmma_bf16<<<gGemm, 256, 0, stream>>>(e_b, wk_b, nullptr, k_b, 1);
  gemm_wmma_bf16<<<gGemm, 256, 0, stream>>>(e_b, wv_b, nullptr, v_b, 1);

  // 4) attention + head-mean (160 KB dynamic LDS: CDNA5 320 KB WGP pool)
  const size_t smemBytes = (16 * Sn + 8 * 16 * HDn) * sizeof(float);
  attention_wmma<<<Bn * (Sn / 16), 256, smemBytes, stream>>>(q_b, k_b, v_b,
                                                             attn_b, mean_out);

  // 5) output projection (WMMA + TDM-staged A), f32 flat
  gemm_wmma_bf16<<<gGemm, 256, 0, stream>>>(attn_b, wo_b, proj, nullptr, 0);

  // 6) residual + bias + LayerNorm
  resid_layernorm<<<TOKENS, 256, 0, stream>>>(e_f32, proj, bo, gamma, beta, out);
}
